// SGATLayer_188978561181
// MI455X (gfx1250) — compile-verified, matrix operand order checked by reading, not run on hardware
//
#include <hip/hip_runtime.h>

#define N_NODES 50000
#define N_EDGES 800000
#define IN_F    256
#define OUT_F   32
#define HEADS   4
#define OUT_FH  128   // OUT_F*HEADS
#define LRELU_ALPHA 0.2f

typedef __attribute__((ext_vector_type(16))) __bf16 v16bf;
typedef __attribute__((ext_vector_type(8)))  float  v8f;

union BV { v16bf v; __bf16 e[16]; };

// Split an fp32 float4 into bf16 hi/lo parts at vector slots [base..base+3].
__device__ __forceinline__ void split_put4(BV& hi, BV& lo, int base, float4 x) {
  float vs[4] = {x.x, x.y, x.z, x.w};
#pragma unroll
  for (int j = 0; j < 4; ++j) {
    __bf16 hb = (__bf16)vs[j];
    hi.e[base + j] = hb;
    lo.e[base + j] = (__bf16)(vs[j] - (float)hb);
  }
}

// K index for 16-bit A/B operand vector slot `idx` given k-half `kh` (ISA 7.12.2).
__device__ __forceinline__ int kmap(int idx, int kh) {
  return (idx & 7) + ((idx >> 3) << 4) + kh * 8;
}

// ---------------- prep kernels (B operands pre-swizzled to fragment layout) ----------------
// Fragment layout: frag[((ntile*8 + ks)*32 + lane)*16 + idx]  -> one 32B chunk per lane.

__global__ void prep_w_swizzle_kernel(const float* __restrict__ W,
                                      __bf16* __restrict__ hi, __bf16* __restrict__ lo) {
  int o = blockIdx.x * blockDim.x + threadIdx.x;      // 8 nt * 8 ks * 32 lane * 16 idx = 32768
  if (o >= 8 * 8 * 32 * 16) return;
  int idx  = o & 15;
  int lane = (o >> 4) & 31;
  int ks   = (o >> 9) & 7;
  int nt   = o >> 12;
  int ml = lane & 15, kh = lane >> 4;
  int K = ks * 32 + kmap(idx, kh);
  int n = nt * 16 + ml;
  float x = W[K * OUT_FH + n];
  __bf16 hb = (__bf16)x;
  hi[o] = hb;
  lo[o] = (__bf16)(x - (float)hb);
}

// B2[K,n] = sum_f W_e[K, n*32+f] * a_edge[n*32+f] for n<4 (zero-padded to 16),
// emitted directly in fragment layout: frag[(ks*32 + lane)*16 + idx].
__global__ void prep_b2_swizzle_kernel(const float* __restrict__ We, const float* __restrict__ a_edge,
                                       __bf16* __restrict__ hi, __bf16* __restrict__ lo) {
  int o = blockIdx.x * blockDim.x + threadIdx.x;      // 8 ks * 32 lane * 16 idx = 4096
  if (o >= 8 * 32 * 16) return;
  int idx  = o & 15;
  int lane = (o >> 4) & 31;
  int ks   = o >> 9;
  int ml = lane & 15, kh = lane >> 4;
  int K = ks * 32 + kmap(idx, kh);
  float s = 0.f;
  if (ml < HEADS) {
    for (int f = 0; f < OUT_F; ++f)
      s += We[K * OUT_FH + ml * OUT_F + f] * a_edge[ml * OUT_F + f];
  }
  __bf16 hb = (__bf16)s;
  hi[o] = hb;
  lo[o] = (__bf16)(s - (float)hb);
}

__global__ void zero_kernel(float4* __restrict__ out, int n4) {
  int i = blockIdx.x * blockDim.x + threadIdx.x;
  if (i < n4) out[i] = make_float4(0.f, 0.f, 0.f, 0.f);
}

__global__ void relu_kernel(float4* __restrict__ out, int n4) {
  int i = blockIdx.x * blockDim.x + threadIdx.x;
  if (i < n4) {
    float4 v = out[i];
    v.x = fmaxf(v.x, 0.f); v.y = fmaxf(v.y, 0.f);
    v.z = fmaxf(v.z, 0.f); v.w = fmaxf(v.w, 0.f);
    out[i] = v;
  }
}

// ---------------- Wh = h @ W  (bf16x3 WMMA, fp32 accumulate) ----------------
// Block = 256 threads = 8 waves; block handles one 16-row M tile, wave w -> N tile w.
__global__ void __launch_bounds__(256)
gemm_node_kernel(const float* __restrict__ h,
                 const __bf16* __restrict__ Whi, const __bf16* __restrict__ Wlo,
                 float* __restrict__ Wh) {
  const int wave = threadIdx.x >> 5;
  const int lane = threadIdx.x & 31;
  const int ml   = lane & 15;    // A: row within tile / B,C,D: column within tile
  const int kh   = lane >> 4;    // K-half select
  const int m0   = blockIdx.x * 16;
  const int n    = wave * 16 + ml;
  const float* arow = h + (size_t)(m0 + ml) * IN_F;

  v8f acc = {};
#pragma unroll
  for (int ks = 0; ks < 8; ++ks) {                 // K = 256 in steps of 32
    const int kb = ks * 32 + kh * 8;
    BV ahi, alo, bhi, blo;
    split_put4(ahi, alo, 0,  *(const float4*)(arow + kb + 0));
    split_put4(ahi, alo, 4,  *(const float4*)(arow + kb + 4));
    split_put4(ahi, alo, 8,  *(const float4*)(arow + kb + 16));
    split_put4(ahi, alo, 12, *(const float4*)(arow + kb + 20));
    const size_t fo = ((size_t)(wave * 8 + ks) * 32 + lane) * 16;
    bhi.v = *(const v16bf*)(Whi + fo);             // 32B contiguous -> 2x b128
    blo.v = *(const v16bf*)(Wlo + fo);
    acc = __builtin_amdgcn_wmma_f32_16x16x32_bf16(false, ahi.v, false, bhi.v, (short)0, acc, false, false);
    acc = __builtin_amdgcn_wmma_f32_16x16x32_bf16(false, ahi.v, false, blo.v, (short)0, acc, false, false);
    acc = __builtin_amdgcn_wmma_f32_16x16x32_bf16(false, alo.v, false, bhi.v, (short)0, acc, false, false);
  }
#pragma unroll
  for (int r = 0; r < 8; ++r) {
    const int m = m0 + r + 8 * kh;                 // C/D layout: VGPR r, lanes16-31 -> M+8
    Wh[(size_t)m * OUT_FH + n] = acc[r];
  }
}

// -------- per-node attention pieces: alpha_src/dst[n,h] = <Wh[n,h,:], a[h,:]> --------
__global__ void __launch_bounds__(256)
node_alpha_kernel(const float* __restrict__ Wh, const float* __restrict__ a_src,
                  const float* __restrict__ a_dst,
                  float* __restrict__ as, float* __restrict__ ad) {
  const int wave = threadIdx.x >> 5, lane = threadIdx.x & 31;
  const int nidx = blockIdx.x * 8 + wave;
  const float* row = Wh + (size_t)nidx * OUT_FH;
#pragma unroll
  for (int hh = 0; hh < HEADS; ++hh) {
    float w = row[hh * OUT_F + lane];
    float s = w * a_src[hh * OUT_F + lane];
    float d = w * a_dst[hh * OUT_F + lane];
#pragma unroll
    for (int off = 16; off > 0; off >>= 1) {
      s += __shfl_xor(s, off, 32);
      d += __shfl_xor(d, off, 32);
    }
    if (lane == 0) { as[nidx * HEADS + hh] = s; ad[nidx * HEADS + hh] = d; }
  }
}

// -------- alpha_e = edge_feat @ B2  (bf16x3 WMMA; N padded 4->16) --------
__global__ void __launch_bounds__(256)
gemm_edge_alpha_kernel(const float* __restrict__ ef,
                       const __bf16* __restrict__ B2hi, const __bf16* __restrict__ B2lo,
                       float* __restrict__ alpha_e) {
  const int wave = threadIdx.x >> 5;
  const int lane = threadIdx.x & 31;
  const int ml = lane & 15, kh = lane >> 4;
  const int etile = blockIdx.x * 8 + wave;
  const int m0 = etile * 16;
  const float* arow = ef + (size_t)(m0 + ml) * IN_F;

  v8f acc = {};
#pragma unroll
  for (int ks = 0; ks < 8; ++ks) {
    const int kb = ks * 32 + kh * 8;
    BV ahi, alo, bhi, blo;
    split_put4(ahi, alo, 0,  *(const float4*)(arow + kb + 0));
    split_put4(ahi, alo, 4,  *(const float4*)(arow + kb + 4));
    split_put4(ahi, alo, 8,  *(const float4*)(arow + kb + 16));
    split_put4(ahi, alo, 12, *(const float4*)(arow + kb + 20));
    const size_t fo = ((size_t)ks * 32 + lane) * 16;
    bhi.v = *(const v16bf*)(B2hi + fo);
    blo.v = *(const v16bf*)(B2lo + fo);
    acc = __builtin_amdgcn_wmma_f32_16x16x32_bf16(false, ahi.v, false, bhi.v, (short)0, acc, false, false);
    acc = __builtin_amdgcn_wmma_f32_16x16x32_bf16(false, ahi.v, false, blo.v, (short)0, acc, false, false);
    acc = __builtin_amdgcn_wmma_f32_16x16x32_bf16(false, alo.v, false, bhi.v, (short)0, acc, false, false);
  }
  if (ml < HEADS) {
#pragma unroll
    for (int r = 0; r < 8; ++r)
      alpha_e[(size_t)(m0 + r + 8 * kh) * HEADS + ml] = acc[r];
  }
}

// -------- per-edge: logits -> leakyrelu -> head-softmax -> scatter-add --------
// One wave per edge; lane l handles output columns 4l..4l+3 (head = l/8).
__global__ void __launch_bounds__(256)
edge_scatter_kernel(const int* __restrict__ ei, const float* __restrict__ Wh,
                    const float* __restrict__ as, const float* __restrict__ ad,
                    const float* __restrict__ ae, float* __restrict__ out) {
  const int wave = threadIdx.x >> 5, lane = threadIdx.x & 31;
  const int e = blockIdx.x * 8 + wave;
  const int src = ei[e];
  const int dst = ei[N_EDGES + e];

  const float4 s4 = *(const float4*)(as + (size_t)src * HEADS);
  const float4 d4 = *(const float4*)(ad + (size_t)dst * HEADS);
  const float4 e4 = *(const float4*)(ae + (size_t)e   * HEADS);

  float logit[HEADS] = {s4.x + d4.x + e4.x, s4.y + d4.y + e4.y,
                        s4.z + d4.z + e4.z, s4.w + d4.w + e4.w};
  float mx = -1e30f;
#pragma unroll
  for (int hh = 0; hh < HEADS; ++hh) {
    float v = logit[hh];
    v = v >= 0.f ? v : LRELU_ALPHA * v;
    logit[hh] = v;
    mx = fmaxf(mx, v);
  }
  float ssum = 0.f;
#pragma unroll
  for (int hh = 0; hh < HEADS; ++hh) { logit[hh] = __expf(logit[hh] - mx); ssum += logit[hh]; }
  const float inv = 1.f / ssum;

  const int c = lane * 4;
  const float att = logit[lane >> 3] * inv;        // head = (4*lane)/32
  float4 w = *(const float4*)(Wh + (size_t)src * OUT_FH + c);
  float* o = out + (size_t)dst * OUT_FH + c;
  unsafeAtomicAdd(o + 0, w.x * att);
  unsafeAtomicAdd(o + 1, w.y * att);
  unsafeAtomicAdd(o + 2, w.z * att);
  unsafeAtomicAdd(o + 3, w.w * att);
}

// ---------------- launch ----------------
extern "C" void kernel_launch(void* const* d_in, const int* in_sizes, int n_in,
                              void* d_out, int out_size, void* d_ws, size_t ws_size,
                              hipStream_t stream) {
  (void)in_sizes; (void)n_in; (void)out_size; (void)ws_size;
  const float* h      = (const float*)d_in[0];
  const float* ef     = (const float*)d_in[1];
  const int*   ei     = (const int*)d_in[2];
  const float* W      = (const float*)d_in[3];
  const float* We     = (const float*)d_in[4];
  const float* a_src  = (const float*)d_in[5];
  const float* a_dst  = (const float*)d_in[6];
  const float* a_edge = (const float*)d_in[7];
  float* out = (float*)d_out;

  // workspace carve-up (~40.2 MB), all segments 16B-aligned
  float* Wh = (float*)d_ws;                                  // [N_NODES, 128]
  float* ae = Wh + (size_t)N_NODES * OUT_FH;                 // [N_EDGES, 4]
  float* as = ae + (size_t)N_EDGES * HEADS;                  // [N_NODES, 4]
  float* ad = as + (size_t)N_NODES * HEADS;                  // [N_NODES, 4]
  __bf16* Whi  = (__bf16*)(ad + (size_t)N_NODES * HEADS);    // swizzled [8][8][32][16]
  __bf16* Wlo  = Whi + IN_F * OUT_FH;
  __bf16* B2hi = Wlo + IN_F * OUT_FH;                        // swizzled [8][32][16]
  __bf16* B2lo = B2hi + IN_F * 16;

  const int n4 = N_NODES * OUT_FH / 4;

  prep_w_swizzle_kernel<<<(IN_F * OUT_FH + 255) / 256, 256, 0, stream>>>(W, Whi, Wlo);
  prep_b2_swizzle_kernel<<<(IN_F * 16 + 255) / 256, 256, 0, stream>>>(We, a_edge, B2hi, B2lo);
  zero_kernel<<<(n4 + 255) / 256, 256, 0, stream>>>((float4*)out, n4);
  gemm_node_kernel<<<N_NODES / 16, 256, 0, stream>>>(h, Whi, Wlo, Wh);
  node_alpha_kernel<<<N_NODES / 8, 256, 0, stream>>>(Wh, a_src, a_dst, as, ad);
  gemm_edge_alpha_kernel<<<N_EDGES / 16 / 8, 256, 0, stream>>>(ef, B2hi, B2lo, ae);
  edge_scatter_kernel<<<N_EDGES / 8, 256, 0, stream>>>(ei, Wh, as, ad, ae, out);
  relu_kernel<<<(n4 + 255) / 256, 256, 0, stream>>>((float4*)out, n4);
}
